// ClusteringLSTM_48704929137378
// MI455X (gfx1250) — matrix operand order repaired and checked by reading, not compiled, not run on hardware
//
#include <hip/hip_runtime.h>
#include <cstdint>
#include <cstddef>

// ---------------------------------------------------------------------------
// Types for CDNA5 WMMA (wave32): v_wmma_f32_16x16x32_bf16
// ---------------------------------------------------------------------------
typedef __bf16 bf16_t;
typedef __attribute__((ext_vector_type(16))) __bf16 v16bf;
typedef __attribute__((ext_vector_type(8)))  __bf16 v8bf;
typedef __attribute__((ext_vector_type(8)))  float  v8f;

constexpr int kB  = 2048;          // sequence length (acts as batch/time)
constexpr int kE  = 256;           // embed dim
constexpr int kH  = 512;           // hidden
constexpr int kV  = 20000;         // vocab
constexpr int kC  = 8;             // clusters
constexpr int kP  = 10;            // num_pred
constexpr int kFH = 4 * kH;        // 2048 gates
constexpr int kK  = 2 * kE;        // 512 == kH (GEMM K dim, both layers)

// d_out layout (floats): loss | preds[B*P] | h[2*H] | c[2*H]
constexpr int OUT_LOSS = 0;
constexpr int OUT_PRED = 1;
constexpr int OUT_H    = OUT_PRED + kB * kP;   // 20481
constexpr int OUT_C    = OUT_H + 2 * kH;       // 21505

__device__ inline v8f wmma_bf16(v16bf a, v16bf b, v8f c) {
  // D = A(16x32 bf16) x B(32x16 bf16) + C(16x16 f32)
  return __builtin_amdgcn_wmma_f32_16x16x32_bf16(
      /*neg_a=*/false, a, /*neg_b=*/false, b,
      /*c_mod=*/(short)0, c, /*reuse_a=*/false, /*reuse_b=*/false);
}

// ---------------------------------------------------------------------------
// Tensor Data Mover (gfx1250): DMA a 2-D fp32 tile (tile_k rows x tile_n cols,
// row stride row_stride elements) from global into LDS.
// This toolchain exposes the 6-arg builtin:
//   (uint32x4 g0, int32x8 g1, int32x4 g2, int32x4 g3, int32x8 g4, i32 cpol)
// ---------------------------------------------------------------------------
#if defined(__gfx1250__) && __has_builtin(__builtin_amdgcn_tensor_load_to_lds) && \
    __has_builtin(__builtin_amdgcn_s_wait_tensorcnt)
#define USE_TDM 1
#else
#define USE_TDM 0
#endif

#if USE_TDM
typedef __attribute__((ext_vector_type(4))) unsigned int v4u;
typedef __attribute__((ext_vector_type(8))) int v8i_t;
typedef __attribute__((ext_vector_type(4))) int v4i_t;

__device__ inline void tdm_load_tile_f32(unsigned lds_addr, const float* gptr,
                                         unsigned rem_n, unsigned rem_k,
                                         unsigned tile_n, unsigned tile_k,
                                         unsigned row_stride) {
  unsigned long long ga = (unsigned long long)(uintptr_t)gptr;
  // D# group 0: count=1 | lds_addr | global_addr[56:0] | type=2 (bits 127:126)
  v4u g0;
  g0[0] = 1u;
  g0[1] = lds_addr;
  g0[2] = (unsigned)ga;
  g0[3] = (unsigned)((ga >> 32) & 0x01FFFFFFu) | 0x80000000u;
  // D# group 1: data_size=4B; tensor_dim0/1 = remaining extent (OOB -> 0-fill);
  // tile_dim0 x tile_dim1; tensor_dim0_stride = row_stride.
  v8i_t g1;
  g1[0] = 0x20000;                                            // data_size=2 (4B)
  g1[1] = (int)((rem_n & 0xFFFFu) << 16);                     // dim0 [63:48]
  g1[2] = (int)((rem_n >> 16) | ((rem_k & 0xFFFFu) << 16));   // dim0 hi | dim1 lo
  g1[3] = (int)((rem_k >> 16) | (tile_n << 16));              // dim1 hi | tile0
  g1[4] = (int)tile_k;                                        // tile1 (tile2=0)
  g1[5] = (int)row_stride;                                    // dim0 stride lo
  g1[6] = 0;                                                  // stride hi, dim1 stride lo
  g1[7] = 0;
  v4i_t z4 = {0, 0, 0, 0};
  v8i_t z8 = {0, 0, 0, 0, 0, 0, 0, 0};
  __builtin_amdgcn_tensor_load_to_lds(g0, g1, z4, z4, z8, 0);
}
#endif

// ---------------------------------------------------------------------------
// Small prep kernels
// ---------------------------------------------------------------------------
__global__ void init_counts_kernel(int* __restrict__ counts) {
  int i = threadIdx.x;
  if (i < kC) counts[i] = 0;
}

__global__ void f32_to_bf16_kernel(const float* __restrict__ src,
                                   bf16_t* __restrict__ dst, int n) {
  int i = blockIdx.x * blockDim.x + threadIdx.x;
  if (i < n) dst[i] = (bf16_t)src[i];
}

// Embedding gather + concat -> X bf16 [B, 2E]
__global__ void embed_kernel(const int* __restrict__ pc, const int* __restrict__ delta,
                             const float* __restrict__ pe, const float* __restrict__ de,
                             bf16_t* __restrict__ X) {
  int idx = blockIdx.x * blockDim.x + threadIdx.x;   // over B*E
  if (idx >= kB * kE) return;
  int b = idx >> 8;            // E = 256
  int e = idx & (kE - 1);
  X[(size_t)b * kK + e]       = (bf16_t)pe[(size_t)pc[b]    * kE + e];
  X[(size_t)b * kK + kE + e]  = (bf16_t)de[(size_t)delta[b] * kE + e];
}

// Per-cluster row gather (compaction): row_list[c*B + slot] = b
__global__ void count_rows_kernel(const int* __restrict__ clusters,
                                  int* __restrict__ counts,
                                  int* __restrict__ row_list) {
  int b = blockIdx.x * blockDim.x + threadIdx.x;
  if (b < kB) {
    int c = clusters[b];
    int slot = atomicAdd(&counts[c], 1);
    row_list[(size_t)c * kB + slot] = b;
  }
}

// ---------------------------------------------------------------------------
// GEMM:  C[M,N] = A[M,K](bf16, row-major) * B[N,K](bf16, row-major)^T, fp32 out
// M=N=2048, K=512. Block 256 threads = 8 waves (2x4), wave tile 32x32,
// WG tile 64x128. Direct global loads into WMMA fragments.
// ---------------------------------------------------------------------------
__global__ __launch_bounds__(256)
void gemm_bf16_nt_kernel(const bf16_t* __restrict__ A,
                         const bf16_t* __restrict__ Bm,
                         float* __restrict__ Cm) {
  const int tid  = threadIdx.x;
  const int lane = tid & 31;
  const int wave = tid >> 5;
  const int wm   = wave >> 2;          // 0..1
  const int wn   = wave & 3;           // 0..3
  const int m0   = blockIdx.y * 64 + wm * 32;
  const int n0   = blockIdx.x * 128 + wn * 32;
  const int hi   = lane >> 4;
  const int l15  = lane & 15;

  const bf16_t* arow0 = A  + (size_t)(m0 + l15)        * kK;
  const bf16_t* arow1 = A  + (size_t)(m0 + 16 + l15)   * kK;
  const bf16_t* brow0 = Bm + (size_t)(n0 + l15)        * kK + 16 * hi;
  const bf16_t* brow1 = Bm + (size_t)(n0 + 16 + l15)   * kK + 16 * hi;

  v8f acc[2][2] = {};
  for (int k0 = 0; k0 < kK; k0 += 32) {
    v16bf a0, a1, b0, b1;
    // A fragment: lanes 0-15 & 16-31 both hold M=l15; K split per half.
    *(v8bf*)&a0       = *(const v8bf*)(arow0 + k0 + 16 * hi);
    *((v8bf*)&a0 + 1) = *(const v8bf*)(arow0 + k0 + 16 * hi + 8);
    *(v8bf*)&a1       = *(const v8bf*)(arow1 + k0 + 16 * hi);
    *((v8bf*)&a1 + 1) = *(const v8bf*)(arow1 + k0 + 16 * hi + 8);
    // B fragment: lane holds column n=l15, K = 16*hi + j, contiguous in W.
    b0 = *(const v16bf*)(brow0 + k0);
    b1 = *(const v16bf*)(brow1 + k0);
    acc[0][0] = wmma_bf16(a0, b0, acc[0][0]);
    acc[0][1] = wmma_bf16(a0, b1, acc[0][1]);
    acc[1][0] = wmma_bf16(a1, b0, acc[1][0]);
    acc[1][1] = wmma_bf16(a1, b1, acc[1][1]);
  }
  // C/D layout: VGPR r -> rows m0 + 8*hi + r, column n0 + l15.
  #pragma unroll
  for (int mt = 0; mt < 2; ++mt)
    #pragma unroll
    for (int nt = 0; nt < 2; ++nt)
      #pragma unroll
      for (int r = 0; r < 8; ++r)
        Cm[(size_t)(m0 + mt * 16 + 8 * hi + r) * kFH + n0 + nt * 16 + l15] =
            acc[mt][nt][r];
}

// ---------------------------------------------------------------------------
// Sequential LSTM layer: gates(t) = G[t] + b_ih + b_hh + Whh @ h(t-1).
// Single workgroup, 512 threads (1 h-index each), h/c live in LDS.
// ---------------------------------------------------------------------------
__global__ __launch_bounds__(512)
void lstm_seq_kernel(const float* __restrict__ G,      // [B, 4H]
                     const float* __restrict__ Whh,    // [4H, H] fp32
                     const float* __restrict__ b_ih,
                     const float* __restrict__ b_hh,
                     const float* __restrict__ h0,     // [H]
                     const float* __restrict__ c0,     // [H]
                     bf16_t* __restrict__ y_bf16,      // [B, H]
                     float* __restrict__ hT_out,       // [H]
                     float* __restrict__ cT_out) {     // [H]
  __shared__ float sh[kH];
  __shared__ float sc[kH];
  const int j = threadIdx.x;
  sh[j] = h0[j];
  sc[j] = c0[j];
  __syncthreads();

  const float* wi = Whh + (size_t)(0 * kH + j) * kH;
  const float* wf = Whh + (size_t)(1 * kH + j) * kH;
  const float* wg = Whh + (size_t)(2 * kH + j) * kH;
  const float* wo = Whh + (size_t)(3 * kH + j) * kH;
  const float bi = b_ih[0 * kH + j] + b_hh[0 * kH + j];
  const float bf = b_ih[1 * kH + j] + b_hh[1 * kH + j];
  const float bg = b_ih[2 * kH + j] + b_hh[2 * kH + j];
  const float bo = b_ih[3 * kH + j] + b_hh[3 * kH + j];

  for (int t = 0; t < kB; ++t) {
    const float* g = G + (size_t)t * kFH;
    float gi = g[0 * kH + j] + bi;
    float gf = g[1 * kH + j] + bf;
    float gg = g[2 * kH + j] + bg;
    float go = g[3 * kH + j] + bo;
    #pragma unroll 4
    for (int k = 0; k < kH; k += 4) {
      float4 vi = *(const float4*)(wi + k);
      float4 vf = *(const float4*)(wf + k);
      float4 vg = *(const float4*)(wg + k);
      float4 vo = *(const float4*)(wo + k);
      float h0v = sh[k], h1v = sh[k + 1], h2v = sh[k + 2], h3v = sh[k + 3];
      gi += vi.x * h0v + vi.y * h1v + vi.z * h2v + vi.w * h3v;
      gf += vf.x * h0v + vf.y * h1v + vf.z * h2v + vf.w * h3v;
      gg += vg.x * h0v + vg.y * h1v + vg.z * h2v + vg.w * h3v;
      go += vo.x * h0v + vo.y * h1v + vo.z * h2v + vo.w * h3v;
    }
    float i_ = 1.0f / (1.0f + __expf(-gi));
    float f_ = 1.0f / (1.0f + __expf(-gf));
    float g_ = tanhf(gg);
    float o_ = 1.0f / (1.0f + __expf(-go));
    float cn = f_ * sc[j] + i_ * g_;
    float hn = o_ * tanhf(cn);
    __syncthreads();
    sc[j] = cn;
    sh[j] = hn;
    y_bf16[(size_t)t * kH + j] = (bf16_t)hn;
    __syncthreads();
  }
  hT_out[j] = sh[j];
  cT_out[j] = sc[j];
}

// ---------------------------------------------------------------------------
// Head GEMM per cluster on gathered rows:
//   logits[row, n] = sum_k Y[row,k] * Wc[c,k,n] + bc[c,n]   (bf16 out)
// Block 256 = 8 waves (2x4). WG tile = 32 gathered rows x 128 cols.
// B-tile path (gfx1250): TENSOR_LOAD_TO_LDS DMAs the fp32 32x128 tile (row
// stride V) into sF; pipeline = wait(i) -> barrier -> convert sF->sB(bf16,
// transposed, stride 40 halves = 20-bank step, conflict-free b128 reads)
// -> barrier -> issue DMA(i+1) -> WMMA on sB, overlapping DMA with compute.
// ---------------------------------------------------------------------------
__global__ __launch_bounds__(256)
void head_gemm_kernel(const bf16_t* __restrict__ Y,      // [B, H] bf16
                      const float* __restrict__ Wc,      // [C, H, V]
                      const float* __restrict__ bc,      // [C, V]
                      const int* __restrict__ row_list,  // [C, B]
                      const int* __restrict__ counts,    // [C]
                      bf16_t* __restrict__ logits) {     // [B, V]
  const int c   = blockIdx.y;
  const int n0  = blockIdx.x * 128;
  const int cnt = counts[c];
  if (cnt == 0) return;
  const float* W    = Wc + (size_t)c * kH * kV;
  const float* bias = bc + (size_t)c * kV;
  const int* rlist  = row_list + (size_t)c * kB;

  __shared__ float  sF[32 * 128];  // fp32 staging tile [k][n] (TDM target)
  __shared__ bf16_t sB[128][40];   // bf16 [n][k], stride 40 halves = 80B

  const int tid  = threadIdx.x;
  const int lane = tid & 31;
  const int wave = tid >> 5;
  const int wm   = wave >> 2;      // 0..1 (row half of WG tile)
  const int wn   = wave & 3;       // 0..3 (32-col slice)
  const int hi   = lane >> 4;
  const int l15  = lane & 15;
  const unsigned rem_n = (unsigned)(kV - n0);

  for (int mbase = 0; mbase < cnt; mbase += 32) {
    const int ia   = mbase + wm * 16 + l15;
    const int rowa = (ia < cnt) ? rlist[ia] : rlist[0];
    const bf16_t* arow = Y + (size_t)rowa * kH;
    v8f acc0 = {}, acc1 = {};

#if USE_TDM
    if (tid < 32)   // wave 0 issues the DMA for the first K tile
      tdm_load_tile_f32((unsigned)(uintptr_t)sF, W + n0, rem_n, (unsigned)kH,
                        128u, 32u, (unsigned)kV);
#endif

    for (int k0 = 0; k0 < kH; k0 += 32) {
#if USE_TDM
      if (tid < 32) __builtin_amdgcn_s_wait_tensorcnt((short)0);
      __syncthreads();                 // publish DMA data; sB readers done
      #pragma unroll
      for (int e = tid; e < 32 * 128; e += 256) {
        int nn = e & 127, kk = e >> 7;
        sB[nn][kk] = (bf16_t)sF[kk * 128 + nn];
      }
      __syncthreads();                 // sF fully consumed -> safe to refill
      if (tid < 32 && k0 + 32 < kH)
        tdm_load_tile_f32((unsigned)(uintptr_t)sF,
                          W + (size_t)(k0 + 32) * kV + n0,
                          rem_n, (unsigned)(kH - (k0 + 32)),
                          128u, 32u, (unsigned)kV);
#else
      __syncthreads();   // protect previous iteration's sB readers
      // Fallback: stage B tile via VGPRs, coalesced over n, LDS transposed.
      #pragma unroll
      for (int e = tid; e < 32 * 128; e += 256) {
        int nn = e & 127, kk = e >> 7;
        int n  = n0 + nn;
        float w = (n < kV) ? W[(size_t)(k0 + kk) * kV + n] : 0.0f;
        sB[nn][kk] = (bf16_t)w;
      }
      if (k0 + 32 < kH) {  // speculative prefetch of next K tile
        int n = n0 + (tid & 127);
        if (n < kV) __builtin_prefetch(W + (size_t)(k0 + 32) * kV + n, 0, 0);
      }
      __syncthreads();
#endif

      v16bf a, b0, b1;
      *(v8bf*)&a       = *(const v8bf*)(arow + k0 + 16 * hi);
      *((v8bf*)&a + 1) = *(const v8bf*)(arow + k0 + 16 * hi + 8);
      const int kb = 16 * hi;
      const int nA = wn * 32 + l15;
      const int nB = nA + 16;
      *(v8bf*)&b0       = *(const v8bf*)&sB[nA][kb];
      *((v8bf*)&b0 + 1) = *(const v8bf*)&sB[nA][kb + 8];
      *(v8bf*)&b1       = *(const v8bf*)&sB[nB][kb];
      *((v8bf*)&b1 + 1) = *(const v8bf*)&sB[nB][kb + 8];
      acc0 = wmma_bf16(a, b0, acc0);
      acc1 = wmma_bf16(a, b1, acc1);
    }

    // Store: frag row r -> gathered row index mbase + wm*16 + 8*hi + r.
    const int na = n0 + wn * 32 + l15;
    const int nb = na + 16;
    #pragma unroll
    for (int r = 0; r < 8; ++r) {
      int im = mbase + wm * 16 + 8 * hi + r;
      if (im < cnt) {
        int row = rlist[im];
        if (na < kV) logits[(size_t)row * kV + na] = (bf16_t)(acc0[r] + bias[na]);
        if (nb < kV) logits[(size_t)row * kV + nb] = (bf16_t)(acc1[r] + bias[nb]);
      }
    }
  }
}

// ---------------------------------------------------------------------------
// Per-row log-softmax NLL + top-10. One WG per row; 20000-entry logit row
// staged in LDS (40 KB -- CDNA5 WGP has 320 KB).
// ---------------------------------------------------------------------------
__global__ __launch_bounds__(256)
void softmax_topk_kernel(const bf16_t* __restrict__ logits,
                         const int* __restrict__ target,
                         float* __restrict__ nll_out,    // [B]
                         float* __restrict__ preds_out)  // [B*P] (as float)
{
  __shared__ bf16_t srow[kV];
  __shared__ float  sred[256];
  __shared__ unsigned long long skey[256];

  const int b   = blockIdx.x;
  const int tid = threadIdx.x;
  const bf16_t* row = logits + (size_t)b * kV;
  for (int v = tid; v < kV; v += 256) srow[v] = row[v];
  __syncthreads();

  // max
  float m = -3.0e38f;
  for (int v = tid; v < kV; v += 256) m = fmaxf(m, (float)srow[v]);
  sred[tid] = m; __syncthreads();
  for (int s = 128; s > 0; s >>= 1) {
    if (tid < s) sred[tid] = fmaxf(sred[tid], sred[tid + s]);
    __syncthreads();
  }
  m = sred[0]; __syncthreads();

  // logsumexp
  float sum = 0.0f;
  for (int v = tid; v < kV; v += 256) sum += __expf((float)srow[v] - m);
  sred[tid] = sum; __syncthreads();
  for (int s = 128; s > 0; s >>= 1) {
    if (tid < s) sred[tid] += sred[tid + s];
    __syncthreads();
  }
  if (tid == 0) {
    float lse = m + __logf(sred[0]);
    nll_out[b] = lse - (float)srow[target[b]];
  }
  __syncthreads();

  // top-10 via repeated argmax (value-desc, index-asc tiebreak)
  for (int p = 0; p < kP; ++p) {
    unsigned long long best = 0ull;
    for (int v = tid; v < kV; v += 256) {
      float f = (float)srow[v];
      unsigned u = __float_as_uint(f);
      u = (u & 0x80000000u) ? ~u : (u | 0x80000000u);   // orderable key
      unsigned long long key =
          ((unsigned long long)u << 32) | (unsigned)(0x7FFFFFFF - v);
      if (key > best) best = key;
    }
    skey[tid] = best; __syncthreads();
    for (int s = 128; s > 0; s >>= 1) {
      if (tid < s && skey[tid + s] > skey[tid]) skey[tid] = skey[tid + s];
      __syncthreads();
    }
    int idx = 0x7FFFFFFF - (int)(skey[0] & 0xFFFFFFFFu);
    if (tid == 0) {
      preds_out[(size_t)b * kP + p] = (float)idx;
      srow[idx] = (bf16_t)(-3.0e38f);   // knock out for next pass
    }
    __syncthreads();
  }
}

// Deterministic loss: per-cluster mean of per-row NLL, summed in fixed order.
__global__ void finalize_kernel(const float* __restrict__ nll,
                                const int* __restrict__ clusters,
                                float* __restrict__ out_loss) {
  if (blockIdx.x != 0 || threadIdx.x != 0) return;
  float sum[kC];
  int   cnt[kC];
  for (int c = 0; c < kC; ++c) { sum[c] = 0.0f; cnt[c] = 0; }
  for (int b = 0; b < kB; ++b) {
    int c = clusters[b];
    sum[c] += nll[b];
    cnt[c] += 1;
  }
  float loss = 0.0f;
  for (int c = 0; c < kC; ++c)
    if (cnt[c] > 0) loss += sum[c] / (float)cnt[c];
  *out_loss = loss;
}

// ---------------------------------------------------------------------------
// Host launcher
// ---------------------------------------------------------------------------
extern "C" void kernel_launch(void* const* d_in, const int* in_sizes, int n_in,
                              void* d_out, int out_size, void* d_ws, size_t ws_size,
                              hipStream_t stream) {
  const int*   pc      = (const int*)  d_in[0];
  const int*   delta   = (const int*)  d_in[1];
  const int*   clusters= (const int*)  d_in[2];
  const int*   target  = (const int*)  d_in[3];
  const float* h0      = (const float*)d_in[4];   // [2, H]
  const float* c0      = (const float*)d_in[5];   // [2, H]
  const float* pe      = (const float*)d_in[6];
  const float* de      = (const float*)d_in[7];
  const float* W_ih0   = (const float*)d_in[8];
  const float* W_hh0   = (const float*)d_in[9];
  const float* b_ih0   = (const float*)d_in[10];
  const float* b_hh0   = (const float*)d_in[11];
  const float* W_ih1   = (const float*)d_in[12];
  const float* W_hh1   = (const float*)d_in[13];
  const float* b_ih1   = (const float*)d_in[14];
  const float* b_hh1   = (const float*)d_in[15];
  const float* Wc      = (const float*)d_in[16];
  const float* bc      = (const float*)d_in[17];
  float* out = (float*)d_out;

  // Workspace carve-up (256B aligned)
  char* wsp = (char*)d_ws;
  auto take = [&](size_t bytes) -> char* {
    char* p = wsp;
    wsp += (bytes + 255) & ~(size_t)255;
    return p;
  };
  bf16_t* Xbf     = (bf16_t*)take((size_t)kB * kK * 2);
  bf16_t* Wih0bf  = (bf16_t*)take((size_t)kFH * kK * 2);
  bf16_t* Wih1bf  = (bf16_t*)take((size_t)kFH * kH * 2);
  float*  G0      = (float*) take((size_t)kB * kFH * 4);
  float*  G1      = (float*) take((size_t)kB * kFH * 4);
  bf16_t* y0bf    = (bf16_t*)take((size_t)kB * kH * 2);
  bf16_t* y1bf    = (bf16_t*)take((size_t)kB * kH * 2);
  int*    counts  = (int*)   take((size_t)kC * 4);
  int*    rowlist = (int*)   take((size_t)kC * kB * 4);
  float*  nll_ws  = (float*) take((size_t)kB * 4);
  bf16_t* logitsw = (bf16_t*)take((size_t)kB * kV * 2);

  // 1) prep: zero counts, convert input-projection weights, embeddings
  init_counts_kernel<<<1, 32, 0, stream>>>(counts);
  f32_to_bf16_kernel<<<(kFH * kK + 255) / 256, 256, 0, stream>>>(W_ih0, Wih0bf, kFH * kK);
  f32_to_bf16_kernel<<<(kFH * kH + 255) / 256, 256, 0, stream>>>(W_ih1, Wih1bf, kFH * kH);
  embed_kernel<<<(kB * kE + 255) / 256, 256, 0, stream>>>(pc, delta, pe, de, Xbf);
  count_rows_kernel<<<(kB + 255) / 256, 256, 0, stream>>>(clusters, counts, rowlist);

  // 2) layer 0: input projection (WMMA) + sequential recurrence
  gemm_bf16_nt_kernel<<<dim3(kFH / 128, kB / 64), 256, 0, stream>>>(Xbf, Wih0bf, G0);
  lstm_seq_kernel<<<1, kH, 0, stream>>>(G0, W_hh0, b_ih0, b_hh0,
                                        h0, c0, y0bf,
                                        out + OUT_H, out + OUT_C);

  // 3) layer 1: input projection (WMMA) + sequential recurrence
  gemm_bf16_nt_kernel<<<dim3(kFH / 128, kB / 64), 256, 0, stream>>>(y0bf, Wih1bf, G1);
  lstm_seq_kernel<<<1, kH, 0, stream>>>(G1, W_hh1, b_ih1, b_hh1,
                                        h0 + kH, c0 + kH, y1bf,
                                        out + OUT_H + kH, out + OUT_C + kH);

  // 4) cluster-gathered head GEMM (WMMA + TDM-staged B tiles) -> bf16 logits
  head_gemm_kernel<<<dim3((kV + 127) / 128, kC), 256, 0, stream>>>(
      y1bf, Wc, bc, rowlist, counts, logitsw);

  // 5) per-row softmax NLL + top-10, then deterministic loss reduction
  softmax_topk_kernel<<<kB, 256, 0, stream>>>(logitsw, target, nll_ws,
                                              out + OUT_PRED);
  finalize_kernel<<<1, 1, 0, stream>>>(nll_ws, clusters, out + OUT_LOSS);
}
